// GPRegression_85461259256475
// MI455X (gfx1250) — matrix operand order repaired and checked by reading, not compiled, not run on hardware
//
#include <hip/hip_runtime.h>

// ---------------------------------------------------------------------------
// GP regression predict, fused:  out = mean + exp(-0.5*sqdist/l^2)*sv @ mu
// sqdist via GEMM (||x||^2 - 2 x.x' + ||x'||^2) on v_wmma_f32_16x16x32_f16
// with an f16 hi/lo split of the fp32 inputs (3 WMMAs per K-chunk, ~2^-22
// relative error). A-stationary: each wave keeps its 16x256 A tile (hi+lo
// fragments) in VGPRs for the entire n-loop. B tiles are streamed into a
// double-buffered LDS tile with global_load_async_to_lds_b128 (ASYNCcnt),
// overlapping the global->LDS copy with WMMA compute.
// The 8192x8192 cov matrix is never materialized.
// Workspace (~16.2 MB): Xt_hi/lo, Xr_hi/lo (f16), xx, yy, partial.
// ---------------------------------------------------------------------------

#define D_DIM 256
#define BM 64             // test rows per block (4 waves x 16)
#define BN 32             // train rows staged per iteration (2 N-tiles/wave)
#define BLOCK_THREADS 128 // 4 waves (wave32)
#define NSPLIT 4          // split of train dimension across blockIdx.y
#define LDB 264           // padded LDS row stride in halfs (528 B, 16B aligned)

typedef __attribute__((ext_vector_type(16))) _Float16 v16h;
typedef __attribute__((ext_vector_type(8)))  float    v8f;

union FragH { v16h v; uint4 q[2]; };
union AccF  { v8f  v; float f[8]; };

// ---- CDNA5 async global->LDS copy (tracked by ASYNCcnt) -------------------
__device__ __forceinline__ void async_b128(const _Float16* gsrc,
                                           const _Float16* lds_dst) {
  // Generic pointer to __shared__ carries the LDS byte offset in its low 32b.
  const uint32_t lds_off = (uint32_t)(uintptr_t)lds_dst;
  asm volatile("global_load_async_to_lds_b128 %0, %1, off"
               :: "v"(lds_off), "v"(gsrc)
               : "memory");
}
__device__ __forceinline__ void wait_async0() {
  asm volatile("s_wait_asynccnt 0" ::: "memory");
}

// B-matrix 32x16 f16 layout: lane l -> column N=l%16; half h=l/16:
//   VGPR0..3 = K 8h..8h+7, VGPR4..7 = K 16+8h..16+8h+7 (matches A)
__device__ __forceinline__ v16h ld_fragB(const _Float16* row, int k0, int h) {
  FragH f;
  f.q[0] = *(const uint4*)(row + k0 + 8 * h);
  f.q[1] = *(const uint4*)(row + k0 + 16 + 8 * h);
  return f.v;
}

// Issue async loads for one 32-row B tile (hi+lo), 16B chunks, whole block.
__device__ __forceinline__ void stage_B(const _Float16* __restrict__ Xr_hi,
                                        const _Float16* __restrict__ Xr_lo,
                                        _Float16 (*Bh)[LDB],
                                        _Float16 (*Bl)[LDB],
                                        int n0, int tid) {
  for (int c = tid; c < BN * 32; c += BLOCK_THREADS) {
    const int r = c >> 5, c8 = c & 31;
    const size_t g = (size_t)(n0 + r) * D_DIM + c8 * 8;
    async_b128(Xr_hi + g, &Bh[r][c8 * 8]);
    async_b128(Xr_lo + g, &Bl[r][c8 * 8]);
  }
}

// ---------------------------------------------------------------------------
// Prep: fp32 -> f16 hi/lo split + row squared norms. One block per row.
// ---------------------------------------------------------------------------
__global__ __launch_bounds__(D_DIM) void prep_split_norm(
    const float* __restrict__ X, _Float16* __restrict__ hi,
    _Float16* __restrict__ lo, float* __restrict__ norm) {
  const int row = blockIdx.x;
  const int t = threadIdx.x;
  const float x = X[row * D_DIM + t];
  const _Float16 h = (_Float16)x;
  const _Float16 l = (_Float16)(x - (float)h);
  hi[row * D_DIM + t] = h;
  lo[row * D_DIM + t] = l;
  float s = x * x;
#pragma unroll
  for (int m = 16; m >= 1; m >>= 1) s += __shfl_xor(s, m, 32);
  __shared__ float red[D_DIM / 32];
  if ((t & 31) == 0) red[t >> 5] = s;
  __syncthreads();
  if (t == 0) {
    float tot = 0.f;
#pragma unroll
    for (int w = 0; w < D_DIM / 32; ++w) tot += red[w];
    norm[row] = tot;
  }
}

// ---------------------------------------------------------------------------
// Fused main kernel. Block: 4 waves, each owns one 16-row test tile (A in
// VGPRs) and computes a 16x32 cov slab per iteration against the shared,
// double-buffered LDS B tile.
// ---------------------------------------------------------------------------
__global__ __launch_bounds__(BLOCK_THREADS) void gp_fused(
    const _Float16* __restrict__ Xt_hi, const _Float16* __restrict__ Xt_lo,
    const _Float16* __restrict__ Xr_hi, const _Float16* __restrict__ Xr_lo,
    const float* __restrict__ xx, const float* __restrict__ yy,
    const float* __restrict__ mu, const float* __restrict__ ls_p,
    const float* __restrict__ sv_p, float* __restrict__ partial,
    int n_train, int n_test) {
  __shared__ _Float16 Bh[2][BN][LDB];
  __shared__ _Float16 Bl[2][BN][LDB];

  const int tid = threadIdx.x;
  const int wave = tid >> 5;
  const int lane = tid & 31;
  const int h = lane >> 4;   // lane half
  const int ln = lane & 15;  // lane-in-half
  const int m0 = blockIdx.x * BM;
  const int slice = blockIdx.y;

  // ---- A-stationary: load this wave's 16x256 hi/lo fragments into VGPRs.
  // A 16x32 f16 layout: lane l -> row M=l%16; half h:
  //   VGPR0..3 = K 8h..8h+7, VGPR4..7 = K 16+8h..16+8h+7
  v16h Ah[8], Al[8];
  {
    const size_t mrow = (size_t)(m0 + wave * 16 + ln) * D_DIM;
    const _Float16* ah = Xt_hi + mrow;
    const _Float16* al = Xt_lo + mrow;
#pragma unroll
    for (int kc = 0; kc < 8; ++kc) {
      FragH f;
      f.q[0] = *(const uint4*)(ah + kc * 32 + 8 * h);
      f.q[1] = *(const uint4*)(ah + kc * 32 + 16 + 8 * h);
      Ah[kc] = f.v;
      f.q[0] = *(const uint4*)(al + kc * 32 + 8 * h);
      f.q[1] = *(const uint4*)(al + kc * 32 + 16 + 8 * h);
      Al[kc] = f.v;
    }
  }

  const float ls = ls_p[0];
  const float sv = sv_p[0];
  const float half_inv_l2 = 0.5f / (ls * ls);

  // C/D layout: lane (h,ln), VGPR j -> element (M = j+8h, N = ln)
  float xxl[8];
#pragma unroll
  for (int j = 0; j < 8; ++j) xxl[j] = xx[m0 + wave * 16 + 8 * h + j];

  float rowacc[8] = {0.f, 0.f, 0.f, 0.f, 0.f, 0.f, 0.f, 0.f};

  const int n_per = n_train / NSPLIT;
  const int n_begin = slice * n_per;
  const int n_end = n_begin + n_per;

  // Prologue: fill buffer 0.
  stage_B(Xr_hi, Xr_lo, Bh[0], Bl[0], n_begin, tid);
  wait_async0();
  __syncthreads();

  int buf = 0;
  for (int n0 = n_begin; n0 < n_end; n0 += BN) {
    // Prefetch next B tile into the other buffer while computing.
    const int nn = n0 + BN;
    if (nn < n_end) stage_B(Xr_hi, Xr_lo, Bh[buf ^ 1], Bl[buf ^ 1], nn, tid);

    const _Float16* b0h = &Bh[buf][ln][0];
    const _Float16* b0l = &Bl[buf][ln][0];
    const _Float16* b1h = &Bh[buf][16 + ln][0];
    const _Float16* b1l = &Bl[buf][16 + ln][0];

    AccF c10, c20, c11, c21;
    const v8f z = {0.f, 0.f, 0.f, 0.f, 0.f, 0.f, 0.f, 0.f};
    c10.v = z; c20.v = z; c11.v = z; c21.v = z;

#pragma unroll
    for (int kc = 0; kc < 8; ++kc) {
      const int k0 = kc * 32;
      const v16h f0h = ld_fragB(b0h, k0, h);
      const v16h f0l = ld_fragB(b0l, k0, h);
      const v16h f1h = ld_fragB(b1h, k0, h);
      const v16h f1l = ld_fragB(b1l, k0, h);
      // cross = hi*hi' + hi*lo' + lo*hi'  (lo*lo' ~ 2^-22, dropped)
      c10.v = __builtin_amdgcn_wmma_f32_16x16x32_f16(false, Ah[kc], false, f0h,
                                                     (short)0, c10.v, false, false);
      c20.v = __builtin_amdgcn_wmma_f32_16x16x32_f16(false, Ah[kc], false, f0l,
                                                     (short)0, c20.v, false, false);
      c20.v = __builtin_amdgcn_wmma_f32_16x16x32_f16(false, Al[kc], false, f0h,
                                                     (short)0, c20.v, false, false);
      c11.v = __builtin_amdgcn_wmma_f32_16x16x32_f16(false, Ah[kc], false, f1h,
                                                     (short)0, c11.v, false, false);
      c21.v = __builtin_amdgcn_wmma_f32_16x16x32_f16(false, Ah[kc], false, f1l,
                                                     (short)0, c21.v, false, false);
      c21.v = __builtin_amdgcn_wmma_f32_16x16x32_f16(false, Al[kc], false, f1h,
                                                     (short)0, c21.v, false, false);
    }

    // Fused RBF + matvec epilogue for both N-tiles (cov never materialized).
    {
      const float yy0 = yy[n0 + ln];
      const float mu0 = mu[n0 + ln];
      const float yy1 = yy[n0 + 16 + ln];
      const float mu1 = mu[n0 + 16 + ln];
#pragma unroll
      for (int j = 0; j < 8; ++j) {
        float sq0 = xxl[j] + yy0 - 2.0f * (c10.f[j] + c20.f[j]);
        float sq1 = xxl[j] + yy1 - 2.0f * (c11.f[j] + c21.f[j]);
        sq0 = fmaxf(sq0, 0.0f);
        sq1 = fmaxf(sq1, 0.0f);
        const float cov0 = sv * __expf(-sq0 * half_inv_l2);
        const float cov1 = sv * __expf(-sq1 * half_inv_l2);
        rowacc[j] = fmaf(cov0, mu0, rowacc[j]);
        rowacc[j] = fmaf(cov1, mu1, rowacc[j]);
      }
    }

    wait_async0();    // our prefetch into buf^1 has landed
    __syncthreads();  // everyone done reading buf and prefetching
    buf ^= 1;
  }

  // Row reduction over the 16 lanes of each half (columns of the tile)
#pragma unroll
  for (int j = 0; j < 8; ++j) {
    float s = rowacc[j];
    s += __shfl_xor(s, 1, 32);
    s += __shfl_xor(s, 2, 32);
    s += __shfl_xor(s, 4, 32);
    s += __shfl_xor(s, 8, 32);
    rowacc[j] = s;
  }
  if (ln == 0) {  // lanes 0 (rows j) and 16 (rows j+8)
    float* dst = partial + (size_t)slice * n_test;
#pragma unroll
    for (int j = 0; j < 8; ++j) dst[m0 + wave * 16 + 8 * h + j] = rowacc[j];
  }
}

// ---------------------------------------------------------------------------
// Deterministic finalize: out = mean_const + sum over NSPLIT partials
// ---------------------------------------------------------------------------
__global__ void gp_finalize(const float* __restrict__ partial,
                            const float* __restrict__ mean_const,
                            float* __restrict__ out, int n_test) {
  const int i = blockIdx.x * blockDim.x + threadIdx.x;
  if (i < n_test) {
    float s = mean_const[0];
#pragma unroll
    for (int k = 0; k < NSPLIT; ++k) s += partial[(size_t)k * n_test + i];
    out[i] = s;
  }
}

extern "C" void kernel_launch(void* const* d_in, const int* in_sizes, int n_in,
                              void* d_out, int out_size, void* d_ws, size_t ws_size,
                              hipStream_t stream) {
  const float* Xtest = (const float*)d_in[0];
  const float* Xtrain = (const float*)d_in[1];
  const float* mu = (const float*)d_in[2];
  const float* mean_const = (const float*)d_in[3];
  const float* lengthscale = (const float*)d_in[4];
  const float* signal_var = (const float*)d_in[5];
  const int n_test = in_sizes[0] / D_DIM;
  const int n_train = in_sizes[1] / D_DIM;

  char* ws = (char*)d_ws;
  size_t off = 0;
  _Float16* Xt_hi = (_Float16*)(ws + off); off += (size_t)n_test * D_DIM * sizeof(_Float16);
  _Float16* Xt_lo = (_Float16*)(ws + off); off += (size_t)n_test * D_DIM * sizeof(_Float16);
  _Float16* Xr_hi = (_Float16*)(ws + off); off += (size_t)n_train * D_DIM * sizeof(_Float16);
  _Float16* Xr_lo = (_Float16*)(ws + off); off += (size_t)n_train * D_DIM * sizeof(_Float16);
  float* xx = (float*)(ws + off); off += (size_t)n_test * sizeof(float);
  float* yy = (float*)(ws + off); off += (size_t)n_train * sizeof(float);
  float* partial = (float*)(ws + off); off += (size_t)NSPLIT * n_test * sizeof(float);

  prep_split_norm<<<n_test, D_DIM, 0, stream>>>(Xtest, Xt_hi, Xt_lo, xx);
  prep_split_norm<<<n_train, D_DIM, 0, stream>>>(Xtrain, Xr_hi, Xr_lo, yy);

  dim3 grid(n_test / BM, NSPLIT);
  gp_fused<<<grid, BLOCK_THREADS, 0, stream>>>(Xt_hi, Xt_lo, Xr_hi, Xr_lo, xx, yy,
                                               mu, lengthscale, signal_var,
                                               partial, n_train, n_test);

  gp_finalize<<<(n_test + 255) / 256, 256, 0, stream>>>(partial, mean_const,
                                                        (float*)d_out, n_test);
}
